// BatchHardTripletLoss_73366631350373
// MI455X (gfx1250) — matrix operand order repaired and checked by reading, not compiled, not run on hardware
//
#include <hip/hip_runtime.h>
#include <cstdint>
#include <cstddef>

#define NROW 8192
#define NDIM 512
#define MARGIN_F 0.3f

typedef __attribute__((ext_vector_type(16))) __bf16 v16bf;
typedef __attribute__((ext_vector_type(8)))  float  v8f;

union Frag { v16bf v; uint4 q[2]; };

__device__ __forceinline__ unsigned short bf16_rne(float f) {
    unsigned u = __float_as_uint(f);
    u += 0x7fffu + ((u >> 16) & 1u);            // round-to-nearest-even
    return (unsigned short)(u >> 16);
}
// order-preserving float <-> uint32 encoding (monotonic, so u32 atomicMin/Max == float min/max)
__device__ __forceinline__ unsigned enc_f32(float x) {
    unsigned u = __float_as_uint(x);
    return (u & 0x80000000u) ? ~u : (u | 0x80000000u);
}
__device__ __forceinline__ float dec_f32(unsigned e) {
    unsigned u = (e & 0x80000000u) ? (e & 0x7fffffffu) : ~e;
    return __uint_as_float(u);
}

// ---------------- 1) row L2-normalize, split into bf16 hi + bf16 lo residual ----------------
__global__ void __launch_bounds__(256)
normalize_split_kernel(const float* __restrict__ emb,
                       unsigned short* __restrict__ ehi,
                       unsigned short* __restrict__ elo)
{
    const int row = blockIdx.x;
    const int tid = threadIdx.x;
    const float* src = emb + (size_t)row * NDIM;

    float x0 = src[tid];
    float x1 = src[tid + 256];

    __shared__ float red[256];
    red[tid] = x0 * x0 + x1 * x1;
    __syncthreads();
    for (int off = 128; off > 0; off >>= 1) {
        if (tid < off) red[tid] += red[tid + off];
        __syncthreads();
    }
    const float rinv = 1.0f / sqrtf(red[0]);

    const size_t base = (size_t)row * NDIM;
    float y0 = x0 * rinv, y1 = x1 * rinv;
    unsigned short h0 = bf16_rne(y0), h1 = bf16_rne(y1);
    ehi[base + tid]       = h0;
    ehi[base + tid + 256] = h1;
    elo[base + tid]       = bf16_rne(y0 - __uint_as_float((unsigned)h0 << 16));
    elo[base + tid + 256] = bf16_rne(y1 - __uint_as_float((unsigned)h1 << 16));
}

// ---------------- 2) init encoded running min/max ----------------
__global__ void __launch_bounds__(256)
init_minmax_kernel(unsigned* __restrict__ posmin, unsigned* __restrict__ negmax)
{
    const int i = blockIdx.x * 256 + threadIdx.x;
    if (i < NROW) {
        posmin[i] = 0xFF800000u;  // enc(+inf): running min of dot over positives
        negmax[i] = 0x007FFFFFu;  // enc(-inf): running max of dot over negatives
    }
}

// ---------------- 3) fused split-bf16 Gram + batch-hard mining ----------------
// One wave computes a 64x64 tile of G = E*E^T (upper triangle only, ty <= tx),
// then mines hardest pos (min dot) / hardest neg (max dot) per row AND per column.
__global__ void __launch_bounds__(32)
gram_mine_kernel(const unsigned short* __restrict__ ehi,
                 const unsigned short* __restrict__ elo,
                 const int* __restrict__ labels,
                 unsigned* __restrict__ posmin,
                 unsigned* __restrict__ negmax)
{
    const int tx = blockIdx.x;   // column tile
    const int ty = blockIdx.y;   // row tile
    if (ty > tx) return;         // symmetry: upper triangle only

    const int base_i = ty * 64;
    const int base_j = tx * 64;
    const unsigned lane = threadIdx.x & 31u;
    const unsigned half = lane >> 4;     // 0: lanes 0-15, 1: lanes 16-31
    const unsigned l15  = lane & 15u;

    __shared__ int rowlab[64];
    rowlab[lane]      = labels[base_i + lane];
    rowlab[lane + 32] = labels[base_i + lane + 32];
    __syncthreads();

    // accumulators: 4x4 sub-tiles of 16x16 f32
    v8f acc[4][4];
#pragma unroll
    for (int a = 0; a < 4; ++a)
#pragma unroll
        for (int b = 0; b < 4; ++b)
#pragma unroll
            for (int e = 0; e < 8; ++e) acc[a][b][e] = 0.0f;

    // K loop: 512 in steps of 32
    for (int k0 = 0; k0 < NDIM; k0 += 32) {
        // A fragments (16x32, row-major E, K-contiguous): lane m<16 gets K+0..7,+16..23; m+16 gets +8..15,+24..31
        Frag ah[4], al[4];
#pragma unroll
        for (int ti = 0; ti < 4; ++ti) {
            const size_t off = (size_t)(base_i + ti * 16 + l15) * NDIM + k0 + half * 8;
            const unsigned short* ph = ehi + off;
            const unsigned short* pl = elo + off;
            ah[ti].q[0] = *(const uint4*)(ph);
            ah[ti].q[1] = *(const uint4*)(ph + 16);
            al[ti].q[0] = *(const uint4*)(pl);
            al[ti].q[1] = *(const uint4*)(pl + 16);
        }
#pragma unroll
        for (int tj = 0; tj < 4; ++tj) {
            // B fragment (32x16): lane n<16 gets K+0..7,+16..23 of row j0+n; n+16 gets +8..15,+24..31
            const size_t off = (size_t)(base_j + tj * 16 + l15) * NDIM + k0 + half * 8;
            Frag bh, bl;
            const unsigned short* ph = ehi + off;
            const unsigned short* pl = elo + off;
            bh.q[0] = *(const uint4*)(ph);
            bh.q[1] = *(const uint4*)(ph + 16);
            bl.q[0] = *(const uint4*)(pl);
            bl.q[1] = *(const uint4*)(pl + 16);
#pragma unroll
            for (int ti = 0; ti < 4; ++ti) {
                v8f c = acc[ti][tj];
                // split-precision: hi*hi + hi*lo + lo*hi  (~fp32 accuracy, fp32 accumulate)
                c = __builtin_amdgcn_wmma_f32_16x16x32_bf16(false, ah[ti].v, false, bh.v, (short)0, c, false, false);
                c = __builtin_amdgcn_wmma_f32_16x16x32_bf16(false, ah[ti].v, false, bl.v, (short)0, c, false, false);
                c = __builtin_amdgcn_wmma_f32_16x16x32_bf16(false, al[ti].v, false, bh.v, (short)0, c, false, false);
                acc[ti][tj] = c;
            }
        }
    }

    // column labels for this lane (one per 16-wide sub-tile)
    int jl[4];
#pragma unroll
    for (int tj = 0; tj < 4; ++tj) jl[tj] = labels[base_j + tj * 16 + l15];

    const float INF = __uint_as_float(0x7F800000u);

    // ---- row-side mining: C/D layout -> VGPR r holds M = r + 8*half, N = lane&15 ----
#pragma unroll
    for (int ti = 0; ti < 4; ++ti) {
#pragma unroll
        for (int r = 0; r < 8; ++r) {
            const int i  = base_i + ti * 16 + r + 8 * (int)half;
            const int li = rowlab[ti * 16 + r + 8 * (int)half];
            float pv = INF, nv = -INF;
#pragma unroll
            for (int tj = 0; tj < 4; ++tj) {
                const int j = base_j + tj * 16 + (int)l15;
                const float d = acc[ti][tj][r];
                if (li == jl[tj]) { if (i != j) pv = fminf(pv, d); }
                else              { nv = fmaxf(nv, d); }
            }
            // reduce across the 16 lanes of this half (xor 1..8 stays within the half)
#pragma unroll
            for (int m = 1; m < 16; m <<= 1) {
                pv = fminf(pv, __shfl_xor(pv, m, 32));
                nv = fmaxf(nv, __shfl_xor(nv, m, 32));
            }
            if (l15 == 0) {
                atomicMin(&posmin[i], enc_f32(pv));
                atomicMax(&negmax[i], enc_f32(nv));
            }
        }
    }

    // ---- column-side mining (symmetric partner), off-diagonal tiles only ----
    if (ty != tx) {
#pragma unroll
        for (int tj = 0; tj < 4; ++tj) {
            const int j  = base_j + tj * 16 + (int)l15;
            const int lj = jl[tj];
            float pv = INF, nv = -INF;
#pragma unroll
            for (int ti = 0; ti < 4; ++ti) {
#pragma unroll
                for (int r = 0; r < 8; ++r) {
                    const int li = rowlab[ti * 16 + r + 8 * (int)half];
                    const float d = acc[ti][tj][r];
                    if (li == lj) pv = fminf(pv, d);   // i != j guaranteed (disjoint ranges)
                    else          nv = fmaxf(nv, d);
                }
            }
            // combine the two halves (lanes L and L+16 hold the same column j)
            pv = fminf(pv, __shfl_xor(pv, 16, 32));
            nv = fmaxf(nv, __shfl_xor(nv, 16, 32));
            if (half == 0) {
                atomicMin(&posmin[j], enc_f32(pv));
                atomicMax(&negmax[j], enc_f32(nv));
            }
        }
    }
}

// ---------------- 4) decode + hinge + mean ----------------
__global__ void __launch_bounds__(1024)
loss_kernel(const unsigned* __restrict__ posmin,
            const unsigned* __restrict__ negmax,
            float* __restrict__ out)
{
    const int tid = threadIdx.x;
    float s = 0.0f;
    for (int i = tid; i < NROW; i += 1024) {
        const float hardest_pos = 1.0f - dec_f32(posmin[i]); // -inf if no positive -> relu 0
        const float hardest_neg = 1.0f - dec_f32(negmax[i]);
        const float l = hardest_pos - hardest_neg + MARGIN_F;
        s += (l > 0.0f) ? l : 0.0f;
    }
    __shared__ float red[1024];
    red[tid] = s;
    __syncthreads();
    for (int off = 512; off > 0; off >>= 1) {
        if (tid < off) red[tid] += red[tid + off];
        __syncthreads();
    }
    if (tid == 0) out[0] = red[0] / (float)NROW;
}

extern "C" void kernel_launch(void* const* d_in, const int* in_sizes, int n_in,
                              void* d_out, int out_size, void* d_ws, size_t ws_size,
                              hipStream_t stream)
{
    const float* emb    = (const float*)d_in[0];
    const int*   labels = (const int*)d_in[1];
    float*       out    = (float*)d_out;

    // workspace layout: e_hi (8 MB) | e_lo (8 MB) | posmin (32 KB) | negmax (32 KB)
    unsigned short* ehi = (unsigned short*)d_ws;
    unsigned short* elo = ehi + (size_t)NROW * NDIM;
    unsigned* posmin    = (unsigned*)(elo + (size_t)NROW * NDIM);
    unsigned* negmax    = posmin + NROW;

    normalize_split_kernel<<<NROW, 256, 0, stream>>>(emb, ehi, elo);
    init_minmax_kernel<<<NROW / 256, 256, 0, stream>>>(posmin, negmax);
    gram_mine_kernel<<<dim3(128, 128), 32, 0, stream>>>(ehi, elo, labels, posmin, negmax);
    loss_kernel<<<1, 1024, 0, stream>>>(posmin, negmax, out);
}
